// DepthAttnLayer_25598005084240
// MI455X (gfx1250) — matrix operand order repaired and checked by reading, not compiled, not run on hardware
//
#include <hip/hip_runtime.h>
#include <math.h>

typedef __attribute__((ext_vector_type(16))) _Float16 v16h;
typedef __attribute__((ext_vector_type(8)))  _Float16 h8;
typedef __attribute__((ext_vector_type(8)))  float    v8f;

#define EMBED 256
#define TGT   32400
#define SRC   16896
#define NINT  32400

__device__ __forceinline__ float4 ld4(const float* p, int i) {
    return ((const float4*)p)[i];
}

// ---------------- fp32 -> f16 conversion (one-time staging) ----------------
__global__ __launch_bounds__(256) void cvt_f32_f16(const float* __restrict__ in,
                                                   _Float16* __restrict__ out, long n) {
    const long i = ((long)blockIdx.x * 256 + threadIdx.x) * 8;
    if (i >= n) return;
    const float4 a = ld4(in + i, 0), b = ld4(in + i, 1);
    h8 r;
    r[0] = (_Float16)a.x; r[1] = (_Float16)a.y; r[2] = (_Float16)a.z; r[3] = (_Float16)a.w;
    r[4] = (_Float16)b.x; r[5] = (_Float16)b.y; r[6] = (_Float16)b.z; r[7] = (_Float16)b.w;
    *(h8*)(out + i) = r;
}

// A fragment (16x32 f16): lane l -> m = l&15, kp = l>>4.
// halves e=0..15 <-> K = e + 16*kp  (one 32B contiguous chunk at row + k + 16*kp)
__device__ __forceinline__ v16h afrag(const _Float16* row, int k, int kp) {
    return *(const v16h*)(row + k + 16 * kp);
}

// B fragment (32x16 f16, B[k][n] = W[n][k]): lane l -> n = l&15, kp = l>>4.
// halves e=0..15 <-> K = e + 16*kp  (one 32B contiguous chunk of W row)
__device__ __forceinline__ v16h bfrag(const _Float16* wrow, int k, int kp) {
    return *(const v16h*)(wrow + k + 16 * kp);
}

__device__ __forceinline__ void epi_store(v8f acc, int mbase, int ncol, int kp,
                                          float bv, float alpha, int relu,
                                          const float* res, float* Cf, _Float16* Ch,
                                          int M, int N) {
#pragma unroll
    for (int r = 0; r < 8; ++r) {
        const int m = mbase + r + 8 * kp;   // C/D layout: VGPR r, lanes>=16 hold M=r+8
        if (m < M) {
            float v = (acc[r] + bv) * alpha;
            if (relu) v = fmaxf(v, 0.f);
            if (res)  v += res[(size_t)m * N + ncol];
            if (Ch)   Ch[(size_t)m * N + ncol] = (_Float16)v;
            else      Cf[(size_t)m * N + ncol] = v;
        }
    }
}

// C[M,N] = epilogue(A[M,K] @ W[N,K]^T + bias) ; one wave per 32x32 output tile
__global__ __launch_bounds__(256) void gemm_wmma_f16(
    const _Float16* __restrict__ A, const _Float16* __restrict__ W,
    const float* __restrict__ bias, const float* __restrict__ res,
    float* __restrict__ Cf, _Float16* __restrict__ Ch,
    int M, int N, int K, float alpha, int relu) {
    const int lane = threadIdx.x & 31;
    const int wave = threadIdx.x >> 5;
    const int tiles_n = N >> 5;
    const long wg = (long)blockIdx.x * 8 + wave;
    const int tm = (int)(wg / tiles_n);
    const int tn = (int)(wg % tiles_n);
    const int m0 = tm * 32, n0 = tn * 32;
    if (m0 >= M) return;

    const int kp  = lane >> 4;
    const int r16 = lane & 15;
    const int ma0 = min(m0 + r16, M - 1);
    const int ma1 = min(m0 + 16 + r16, M - 1);
    const _Float16* ar0 = A + (size_t)ma0 * K;
    const _Float16* ar1 = A + (size_t)ma1 * K;
    const _Float16* wr0 = W + (size_t)(n0 + r16) * K;
    const _Float16* wr1 = W + (size_t)(n0 + 16 + r16) * K;

    v8f acc00 = {}, acc01 = {}, acc10 = {}, acc11 = {};
    for (int k = 0; k < K; k += 32) {
        const v16h a0 = afrag(ar0, k, kp);
        const v16h a1 = afrag(ar1, k, kp);
        const v16h b0 = bfrag(wr0, k, kp);
        const v16h b1 = bfrag(wr1, k, kp);
        acc00 = __builtin_amdgcn_wmma_f32_16x16x32_f16(false, a0, false, b0, (short)0, acc00, false, false);
        acc01 = __builtin_amdgcn_wmma_f32_16x16x32_f16(false, a0, false, b1, (short)0, acc01, false, false);
        acc10 = __builtin_amdgcn_wmma_f32_16x16x32_f16(false, a1, false, b0, (short)0, acc10, false, false);
        acc11 = __builtin_amdgcn_wmma_f32_16x16x32_f16(false, a1, false, b1, (short)0, acc11, false, false);
    }
    const float bv0 = bias[n0 + r16];
    const float bv1 = bias[n0 + 16 + r16];
    epi_store(acc00, m0,      n0 + r16,      kp, bv0, alpha, relu, res, Cf, Ch, M, N);
    epi_store(acc01, m0,      n0 + 16 + r16, kp, bv1, alpha, relu, res, Cf, Ch, M, N);
    epi_store(acc10, m0 + 16, n0 + r16,      kp, bv0, alpha, relu, res, Cf, Ch, M, N);
    epi_store(acc11, m0 + 16, n0 + 16 + r16, kp, bv1, alpha, relu, res, Cf, Ch, M, N);
}

// Ragged gather-attention with online (streaming) softmax.
// One wave per interval; lane l owns elements [8l, 8l+8) of the 256-wide row
// (head h = l>>2, dims (l&3)*8 .. +7). 4-lane groups share one head's logit.
// k/v rows gathered as f16 (halves L2 gather traffic); math in fp32.
// Output written as f16 (it only feeds the out-proj GEMM).
__global__ __launch_bounds__(256) void depth_attn_ragged(
    const float* __restrict__ qp, const _Float16* __restrict__ k16,
    const _Float16* __restrict__ v16, const int* __restrict__ rfeat,
    const int* __restrict__ istart, const int* __restrict__ ilen,
    _Float16* __restrict__ out16) {
    const int lane = threadIdx.x & 31;
    const int wave = threadIdx.x >> 5;
    const int iv = blockIdx.x * 8 + wave;
    if (iv >= NINT) return;

    const int start = istart[iv];
    const int len   = ilen[iv];
    const int off   = lane * 8;

    const float4* qv = (const float4*)(qp + (size_t)iv * EMBED + off);
    const float4 q0 = qv[0], q1 = qv[1];
    float q[8];
    q[0] = q0.x; q[1] = q0.y; q[2] = q0.z; q[3] = q0.w;
    q[4] = q1.x; q[5] = q1.y; q[6] = q1.z; q[7] = q1.w;

    float m = -INFINITY, s = 0.f;
    float a[8];
#pragma unroll
    for (int i = 0; i < 8; ++i) a[i] = 0.f;

    if (len > 0) {
        // software-pipelined gather: next point's rows load under current math
        int f = rfeat[start];
        h8 kc = *(const h8*)(k16 + (size_t)f * EMBED + off);
        h8 vc = *(const h8*)(v16 + (size_t)f * EMBED + off);
        for (int t = 0; t < len; ++t) {
            h8 kn, vn;
            if (t + 1 < len) {
                const int fn = rfeat[start + t + 1];
                kn = *(const h8*)(k16 + (size_t)fn * EMBED + off);
                vn = *(const h8*)(v16 + (size_t)fn * EMBED + off);
            }
            float e = 0.f;
#pragma unroll
            for (int i = 0; i < 8; ++i) e += q[i] * (float)kc[i];
            e += __shfl_xor(e, 1, 32);   // reduce across the 4-lane head group
            e += __shfl_xor(e, 2, 32);

            const float mn = fmaxf(m, e);
            const float c  = __expf(m - mn);   // == 0 on first point (m = -inf)
            const float p  = __expf(e - mn);
            s = s * c + p;
#pragma unroll
            for (int i = 0; i < 8; ++i) a[i] = a[i] * c + p * (float)vc[i];
            m = mn;
            kc = kn; vc = vn;
        }
    }

    const float inv = (s > 0.f) ? (1.f / s) : 0.f;
    h8 o;
#pragma unroll
    for (int i = 0; i < 8; ++i) o[i] = (_Float16)(a[i] * inv);
    *(h8*)(out16 + (size_t)iv * EMBED + off) = o;
}

// y = LayerNorm(x0 + dx) * w + b ; dual output fp32 (residual) + f16 (GEMM input)
__global__ __launch_bounds__(256) void resid_layernorm(
    const float* __restrict__ x0, const float* __restrict__ dx,
    const float* __restrict__ w, const float* __restrict__ b,
    float* __restrict__ y, _Float16* __restrict__ y16) {
    const int lane = threadIdx.x & 31;
    const int wave = threadIdx.x >> 5;
    const int row = blockIdx.x * 8 + wave;
    if (row >= TGT) return;
    const int off = lane * 8;

    const float4* p0 = (const float4*)(x0 + (size_t)row * EMBED + off);
    const float4* p1 = (const float4*)(dx + (size_t)row * EMBED + off);
    float v[8];
    {
        float4 a0 = p0[0], a1 = p0[1], b0 = p1[0], b1 = p1[1];
        v[0] = a0.x + b0.x; v[1] = a0.y + b0.y; v[2] = a0.z + b0.z; v[3] = a0.w + b0.w;
        v[4] = a1.x + b1.x; v[5] = a1.y + b1.y; v[6] = a1.z + b1.z; v[7] = a1.w + b1.w;
    }
    float sum = 0.f, sq = 0.f;
#pragma unroll
    for (int i = 0; i < 8; ++i) { sum += v[i]; sq += v[i] * v[i]; }
#pragma unroll
    for (int d = 1; d < 32; d <<= 1) {
        sum += __shfl_xor(sum, d, 32);
        sq  += __shfl_xor(sq,  d, 32);
    }
    const float mu   = sum * (1.f / EMBED);
    const float var  = sq * (1.f / EMBED) - mu * mu;
    const float rstd = rsqrtf(var + 1e-5f);

    float o[8];
    const float* wp = w + off;
    const float* bp = b + off;
#pragma unroll
    for (int i = 0; i < 8; ++i) o[i] = (v[i] - mu) * rstd * wp[i] + bp[i];

    float4 o0, o1;
    o0.x = o[0]; o0.y = o[1]; o0.z = o[2]; o0.w = o[3];
    o1.x = o[4]; o1.y = o[5]; o1.z = o[6]; o1.w = o[7];
    float4* ov = (float4*)(y + (size_t)row * EMBED + off);
    ov[0] = o0; ov[1] = o1;

    h8 oh;
#pragma unroll
    for (int i = 0; i < 8; ++i) oh[i] = (_Float16)o[i];
    *(h8*)(y16 + (size_t)row * EMBED + off) = oh;
}

static inline void launch_cvt(const float* in, _Float16* out, long n, hipStream_t s) {
    hipLaunchKernelGGL(cvt_f32_f16, dim3((unsigned)((n / 8 + 255) / 256)), dim3(256), 0, s, in, out, n);
}

static inline void launch_gemm(const _Float16* A, const _Float16* W, const float* bias,
                               const float* res, float* Cf, _Float16* Ch,
                               int M, int N, int K, float alpha, int relu, hipStream_t s) {
    const int tiles = ((M + 31) / 32) * (N / 32);
    const int blocks = (tiles + 7) / 8;
    hipLaunchKernelGGL(gemm_wmma_f16, dim3(blocks), dim3(256), 0, s,
                       A, W, bias, res, Cf, Ch, M, N, K, alpha, relu);
}

extern "C" void kernel_launch(void* const* d_in, const int* in_sizes, int n_in,
                              void* d_out, int out_size, void* d_ws, size_t ws_size,
                              hipStream_t stream) {
    const float* query = (const float*)d_in[0];
    const float* key   = (const float*)d_in[1];
    const float* value = (const float*)d_in[2];
    const float* ipw   = (const float*)d_in[3];   // [768,256]
    const float* ipb   = (const float*)d_in[4];   // [768]
    const float* opw   = (const float*)d_in[5];   // [256,256]
    const float* opb   = (const float*)d_in[6];
    const float* n1w   = (const float*)d_in[7];
    const float* n1b   = (const float*)d_in[8];
    const float* w1    = (const float*)d_in[9];   // [512,256]
    const float* b1    = (const float*)d_in[10];
    const float* w2    = (const float*)d_in[11];  // [256,512]
    const float* b2    = (const float*)d_in[12];
    const int* rfeat   = (const int*)d_in[13];
    const int* istart  = (const int*)d_in[15];
    const int* ilen    = (const int*)d_in[16];
    float* out = (float*)d_out;

    // ---- workspace layout (256B aligned chunks) ----
    char* p = (char*)d_ws;
    auto alloc = [&](size_t bytes) { void* r = (void*)p; p += (bytes + 255) & ~(size_t)255; return r; };
    _Float16* key16   = (_Float16*)alloc((size_t)SRC * EMBED * 2);
    _Float16* query16 = (_Float16*)alloc((size_t)TGT * EMBED * 2);
    _Float16* value16 = (_Float16*)alloc((size_t)SRC * EMBED * 2);
    _Float16* wk16    = (_Float16*)alloc((size_t)EMBED * EMBED * 2);
    _Float16* wq16    = (_Float16*)alloc((size_t)EMBED * EMBED * 2);
    _Float16* wo16    = (_Float16*)alloc((size_t)EMBED * EMBED * 2);
    _Float16* w116    = (_Float16*)alloc((size_t)2 * EMBED * EMBED * 2);
    _Float16* w216    = (_Float16*)alloc((size_t)2 * EMBED * EMBED * 2);
    _Float16* kproj16 = (_Float16*)alloc((size_t)SRC * EMBED * 2);
    float*    qproj   = (float*)alloc((size_t)TGT * EMBED * 4);
    _Float16* attno16 = (_Float16*)alloc((size_t)TGT * EMBED * 2);
    float*    xln     = (float*)alloc((size_t)TGT * EMBED * 4);
    _Float16* h116    = (_Float16*)alloc((size_t)TGT * 2 * EMBED * 2);
    float*    attnp   = qproj;     // reuse: qproj dead after attention
    _Float16* x16     = attno16;   // reuse: attn-out16 dead after out-proj

    const float scaling = 0.17677669529663687f;  // 32^-0.5

    // one-time f16 staging of activations, value table and weights
    launch_cvt(key,   key16,   (long)SRC * EMBED, stream);
    launch_cvt(query, query16, (long)TGT * EMBED, stream);
    launch_cvt(value, value16, (long)SRC * EMBED, stream);
    launch_cvt(ipw,                             wk16, (long)EMBED * EMBED, stream);
    launch_cvt(ipw + (size_t)2 * EMBED * EMBED, wq16, (long)EMBED * EMBED, stream);
    launch_cvt(opw,                             wo16, (long)EMBED * EMBED, stream);
    launch_cvt(w1,                              w116, (long)2 * EMBED * EMBED, stream);
    launch_cvt(w2,                              w216, (long)2 * EMBED * EMBED, stream);

    // k = key @ Wk^T + bk   (stored f16: it's only gathered by the attention stage)
    launch_gemm(key16, wk16, ipb, nullptr, nullptr, kproj16, SRC, EMBED, EMBED, 1.f, 0, stream);
    // q = (query @ Wq^T + bq) * scaling   (fp32: read once per interval)
    launch_gemm(query16, wq16, ipb + 2 * EMBED, nullptr, qproj, nullptr, TGT, EMBED, EMBED, scaling, 0, stream);
    // ragged interval softmax-attention (value = raw value tensor), f16 in/out
    hipLaunchKernelGGL(depth_attn_ragged, dim3((NINT + 7) / 8), dim3(256), 0, stream,
                       qproj, kproj16, value16, rfeat, istart, ilen, attno16);
    // attn = out @ Wo^T + bo
    launch_gemm(attno16, wo16, opb, nullptr, attnp, nullptr, TGT, EMBED, EMBED, 1.f, 0, stream);
    // x = LN(query + attn)  -> fp32 (residual) + f16 (GEMM input)
    hipLaunchKernelGGL(resid_layernorm, dim3((TGT + 7) / 8), dim3(256), 0, stream,
                       query, attnp, n1w, n1b, xln, x16);
    // h1 = relu(x @ W1^T + b1), stored f16
    launch_gemm(x16, w116, b1, nullptr, nullptr, h116, TGT, 2 * EMBED, EMBED, 1.f, 1, stream);
    // out = x + (h1 @ W2^T + b2)
    launch_gemm(h116, w216, b2, xln, out, nullptr, TGT, EMBED, 2 * EMBED, 1.f, 0, stream);
}
